// GATForPPA_78864189489395
// MI455X (gfx1250) — matrix operand order repaired and checked by reading, not compiled, not run on hardware
//
#include <hip/hip_runtime.h>
#include <hip/hip_bf16.h>

// ---------------------------------------------------------------------------
// MI455X (gfx1250) implementation.
//  - Dense matmuls (x@Win, h@Wgat) via v_wmma_f32_16x16x32_f16 (f32->f16 in,
//    f32 accumulate). 16x16 C tile per wave, 8 waves/block cover the 128 cols.
//  - Edge softmax-aggregation: 3 passes (atomic max, exp + atomic-add of
//    numerator/denominator, normalize). Node arrays (51MB each) are L2
//    resident (192MB L2), so random gathers/atomics stay on-chip.
//  - wave32 reductions with __shfl_xor width 32 for layernorm / dot products.
// ---------------------------------------------------------------------------

typedef __attribute__((ext_vector_type(16))) _Float16 v16h;
typedef __attribute__((ext_vector_type(8)))  float    v8f;

#define HID_ 128

// ---------------- float atomic max (int-punning, portable) -----------------
__device__ inline void atomicMaxFloat(float* addr, float v) {
  if (v >= 0.0f) atomicMax((int*)addr, __float_as_int(v));
  else           atomicMin((unsigned int*)addr, (unsigned int)__float_as_int(v));
}

// ---------------- fill kernel (for -inf init etc.) -------------------------
__global__ void fill_kernel(float* p, float v, long long n) {
  long long i = (long long)blockIdx.x * blockDim.x + threadIdx.x;
  if (i < n) p[i] = v;
}

// ---------------- WMMA GEMM: C[M,128] = A[M,K] @ W[K,128] (+bias) ----------
// ISA 16-bit A 16x32 layout: lanes 0-15 row M=lane K-groups {0..7,16..23},
// lanes 16-31 same rows, K-groups {8..15,24..31}. B 32x16: rows K in the same groups as A
// (lanes 0-15 / 16-31), col = lane&15.
// C f32: VGPR r -> row r + 8*(lane>>4), col lane&15.
template<int K>
__global__ __launch_bounds__(256)
void gemm_n128_wmma(const float* __restrict__ A, const float* __restrict__ W,
                    const float* __restrict__ bias, float* __restrict__ C, int M)
{
  const int lane = threadIdx.x & 31;
  const int wv   = threadIdx.x >> 5;          // 0..7 -> 16-col tile
  const int row0 = blockIdx.x * 16;
  if (row0 >= M) return;
  int arow = row0 + (lane & 15);
  if (arow >= M) arow = M - 1;
  const int kgrp = (lane >> 4) * 8;           // A K-group base: 0 or 8
  const int bk   = (lane >> 4) * 8;           // B K-group base: 0 or 8
  const int col  = wv * 16 + (lane & 15);

  v8f acc = {};
  for (int kk = 0; kk < K; kk += 32) {
    v16h a, b;
#pragma unroll
    for (int i = 0; i < 16; ++i) {
      const int j = i >> 1, p = i & 1;
      const int ka = kk + ((j < 4) ? 0 : 16) + kgrp + ((j & 3) * 2 + p);
      a[i] = (_Float16)A[(long long)arow * K + ka];
    }
#pragma unroll
    for (int i = 0; i < 16; ++i) {
      const int j = i >> 1, p = i & 1;
      const int kb = kk + ((j < 4) ? 0 : 16) + bk + ((j & 3) * 2 + p);
      b[i] = (_Float16)W[kb * HID_ + col];
    }
    acc = __builtin_amdgcn_wmma_f32_16x16x32_f16(
        false, a, false, b, (short)0, acc, false, false);
  }
  const float bv = bias ? bias[col] : 0.0f;
  const int crow = row0 + (lane >> 4) * 8;
#pragma unroll
  for (int r = 0; r < 8; ++r) {
    const int rr = crow + r;
    if (rr < M) C[(long long)rr * HID_ + col] = acc[r] + bv;
  }
}

// ---------------- edge MLP -> crit[dst] ------------------------------------
__global__ void edge_mlp_kernel(const float* __restrict__ ea,
                                const int* __restrict__ dst,
                                const float* __restrict__ w1, const float* __restrict__ b1,
                                const float* __restrict__ w2, const float* __restrict__ b2,
                                float* __restrict__ crit, int E_)
{
  int e = blockIdx.x * blockDim.x + threadIdx.x;
  if (e >= E_) return;
  const float a0 = ea[e * 2 + 0], a1 = ea[e * 2 + 1];
  float s = 0.0f;
#pragma unroll
  for (int hh = 0; hh < 16; ++hh) {
    float v = fmaxf(a0 * w1[0 * 16 + hh] + a1 * w1[1 * 16 + hh] + b1[hh], 0.0f);
    float c = w2[hh * 4 + 0] + w2[hh * 4 + 1] + w2[hh * 4 + 2] + w2[hh * 4 + 3];
    s += v * c;
  }
  s = s * 0.25f + 0.25f * (b2[0] + b2[1] + b2[2] + b2[3]);
  atomicAdd(&crit[dst[e]], s);
}

// ---------------- per-node attention coefficients a_s/a_d ------------------
__global__ __launch_bounds__(256)
void attn_coeff_kernel(const float* __restrict__ hp, const float* __restrict__ att_s,
                       const float* __restrict__ att_d, float* __restrict__ a_s,
                       float* __restrict__ a_d, int Nn)
{
  const int lane = threadIdx.x & 31;
  const int node = (int)(((long long)blockIdx.x * blockDim.x + threadIdx.x) >> 5);
  if (node >= Nn) return;
#pragma unroll
  for (int hh = 0; hh < 4; ++hh) {
    const float v = hp[(long long)node * HID_ + hh * 32 + lane];
    float ss = v * att_s[hh * 32 + lane];
    float dd = v * att_d[hh * 32 + lane];
#pragma unroll
    for (int off = 16; off; off >>= 1) {
      ss += __shfl_xor(ss, off, 32);
      dd += __shfl_xor(dd, off, 32);
    }
    if (lane == 0) { a_s[node * 4 + hh] = ss; a_d[node * 4 + hh] = dd; }
  }
}

// ---------------- edge pass 1: segment max of leaky-relu logits ------------
__global__ void edge_max_kernel(const int* __restrict__ src, const int* __restrict__ dst,
                                const float* __restrict__ a_s, const float* __restrict__ a_d,
                                float* __restrict__ m, int E_, int Nn)
{
  int e = blockIdx.x * blockDim.x + threadIdx.x;
  if (e >= E_ + Nn) return;
  const int s = (e < E_) ? src[e] : (e - E_);
  const int d = (e < E_) ? dst[e] : (e - E_);
#pragma unroll
  for (int hh = 0; hh < 4; ++hh) {
    float v = a_s[s * 4 + hh] + a_d[d * 4 + hh];
    v = (v > 0.0f) ? v : 0.2f * v;
    atomicMaxFloat(&m[d * 4 + hh], v);
  }
}

// ---------------- edge pass 2: exp-weighted scatter accumulate -------------
// one thread per (edge, channel-in-head); handles all 4 heads at channel c.
__global__ void edge_accum_kernel(const int* __restrict__ src, const int* __restrict__ dst,
                                  const float* __restrict__ a_s, const float* __restrict__ a_d,
                                  const float* __restrict__ m, const float* __restrict__ hp,
                                  float* __restrict__ hacc, float* __restrict__ denom,
                                  int E_, int Nn)
{
  const long long t = (long long)blockIdx.x * blockDim.x + threadIdx.x;
  const int lane = (int)(t & 31);
  const long long e = t >> 5;
  if (e >= (long long)E_ + Nn) return;
  const int s = (e < E_) ? src[e] : (int)(e - E_);
  const int d = (e < E_) ? dst[e] : (int)(e - E_);
#pragma unroll
  for (int hh = 0; hh < 4; ++hh) {
    float v = a_s[s * 4 + hh] + a_d[d * 4 + hh];
    v = (v > 0.0f) ? v : 0.2f * v;
    const float ex = __expf(v - m[d * 4 + hh]);
    if (lane == 0) atomicAdd(&denom[d * 4 + hh], ex);
    atomicAdd(&hacc[(long long)d * HID_ + hh * 32 + lane],
              ex * hp[(long long)s * HID_ + hh * 32 + lane]);
  }
}

// ---------------- node update: normalize, +bias, layernorm, +crit, resid ---
__global__ __launch_bounds__(256)
void node_update_kernel(const float* __restrict__ hacc, const float* __restrict__ denom,
                        const float* __restrict__ bias, const float* __restrict__ g,
                        const float* __restrict__ bb, const float* __restrict__ crit,
                        float* __restrict__ h, int Nn)
{
  const int lane = threadIdx.x & 31;
  const int node = (int)(((long long)blockIdx.x * blockDim.x + threadIdx.x) >> 5);
  if (node >= Nn) return;
  float v[4];
  float sum = 0.0f, sq = 0.0f;
#pragma unroll
  for (int hh = 0; hh < 4; ++hh) {
    const int c = hh * 32 + lane;
    const float x = hacc[(long long)node * HID_ + c] / denom[node * 4 + hh] + bias[c];
    v[hh] = x; sum += x; sq += x * x;
  }
#pragma unroll
  for (int off = 16; off; off >>= 1) {
    sum += __shfl_xor(sum, off, 32);
    sq  += __shfl_xor(sq,  off, 32);
  }
  const float mu  = sum * (1.0f / HID_);
  const float var = sq * (1.0f / HID_) - mu * mu;
  const float rs  = rsqrtf(var + 1e-5f);
  const float cr  = crit[node];
#pragma unroll
  for (int hh = 0; hh < 4; ++hh) {
    const int c = hh * 32 + lane;
    const long long idx = (long long)node * HID_ + c;
    h[idx] = (v[hh] - mu) * rs * g[c] + bb[c] + cr + h[idx];
  }
}

// ---------------- global attention gate + per-batch max --------------------
__global__ __launch_bounds__(256)
void gate_kernel(const float* __restrict__ h, const float* __restrict__ w1,
                 const float* __restrict__ b1, const float* __restrict__ w2,
                 const float* __restrict__ b2, const int* __restrict__ batch,
                 float* __restrict__ gate, float* __restrict__ gm, int Nn)
{
  const int lane = threadIdx.x & 31;
  const int node = (int)(((long long)blockIdx.x * blockDim.x + threadIdx.x) >> 5);
  if (node >= Nn) return;
  float acc = 0.0f;
#pragma unroll
  for (int rep = 0; rep < 2; ++rep) {
    const int j = lane + rep * 32;      // hidden unit 0..63
    float s = b1[j];
    for (int c = 0; c < HID_; ++c) s += h[(long long)node * HID_ + c] * w1[c * 64 + j];
    acc += tanhf(s) * w2[j];
  }
#pragma unroll
  for (int off = 16; off; off >>= 1) acc += __shfl_xor(acc, off, 32);
  if (lane == 0) {
    const float gv = acc + b2[0];
    gate[node] = gv;
    atomicMaxFloat(&gm[batch[node]], gv);
  }
}

// ---------------- attention pool accumulate --------------------------------
__global__ void pool_accum_kernel(const float* __restrict__ h, const float* __restrict__ gate,
                                  const float* __restrict__ gm, const int* __restrict__ batch,
                                  float* __restrict__ gs, float* __restrict__ gemb, int Nn)
{
  const long long t = (long long)blockIdx.x * blockDim.x + threadIdx.x;
  const int lane = (int)(t & 31);
  const int node = (int)(t >> 5);
  if (node >= Nn) return;
  const int b = batch[node];
  const float ge = __expf(gate[node] - gm[b]);
  if (lane == 0) atomicAdd(&gs[b], ge);
#pragma unroll
  for (int hh = 0; hh < 4; ++hh) {
    const int c = hh * 32 + lane;
    atomicAdd(&gemb[b * HID_ + c], ge * h[(long long)node * HID_ + c]);
  }
}

// ---------------- spatial CNN ----------------------------------------------
__global__ void conv1_kernel(const float* __restrict__ in, const float* __restrict__ w,
                             const float* __restrict__ bias, float* __restrict__ out, int total)
{
  int idx = blockIdx.x * blockDim.x + threadIdx.x;
  if (idx >= total) return;                 // [B,16,64,64]
  const int x = idx & 63, y = (idx >> 6) & 63, co = (idx >> 12) & 15, b = idx >> 16;
  float s = bias[co];
#pragma unroll
  for (int ky = 0; ky < 5; ++ky) {
    const int iy = 2 * y + ky - 2;
    if (iy < 0 || iy >= 128) continue;
#pragma unroll
    for (int kx = 0; kx < 5; ++kx) {
      const int ix = 2 * x + kx - 2;
      if (ix < 0 || ix >= 128) continue;
      s += in[(b * 128 + iy) * 128 + ix] * w[(co * 5 + ky) * 5 + kx];
    }
  }
  out[idx] = fmaxf(s, 0.0f);
}

__global__ void conv2_kernel(const float* __restrict__ in, const float* __restrict__ w,
                             const float* __restrict__ bias, float* __restrict__ out, int total)
{
  int idx = blockIdx.x * blockDim.x + threadIdx.x;
  if (idx >= total) return;                 // [B,32,32,32]
  const int x = idx & 31, y = (idx >> 5) & 31, co = (idx >> 10) & 31, b = idx >> 15;
  float s = bias[co];
  for (int ci = 0; ci < 16; ++ci) {
#pragma unroll
    for (int ky = 0; ky < 5; ++ky) {
      const int iy = 2 * y + ky - 2;
      if (iy < 0 || iy >= 64) continue;
#pragma unroll
      for (int kx = 0; kx < 5; ++kx) {
        const int ix = 2 * x + kx - 2;
        if (ix < 0 || ix >= 64) continue;
        s += in[((b * 16 + ci) * 64 + iy) * 64 + ix] * w[((co * 16 + ci) * 5 + ky) * 5 + kx];
      }
    }
  }
  out[idx] = fmaxf(s, 0.0f);
}

__global__ void avgpool_kernel(const float* __restrict__ in, float* __restrict__ out, int total)
{
  int idx = blockIdx.x * blockDim.x + threadIdx.x;
  if (idx >= total) return;                 // [B,32,4,4]
  const int px = idx & 3, py = (idx >> 2) & 3, c = (idx >> 4) & 31, b = idx >> 9;
  float s = 0.0f;
  for (int dy = 0; dy < 8; ++dy)
    for (int dx = 0; dx < 8; ++dx)
      s += in[((b * 32 + c) * 32 + py * 8 + dy) * 32 + px * 8 + dx];
  out[idx] = s * (1.0f / 64.0f);
}

__global__ void sp_fc_kernel(const float* __restrict__ pooled, const float* __restrict__ w,
                             const float* __restrict__ bias, float* __restrict__ semb, int total)
{
  int idx = blockIdx.x * blockDim.x + threadIdx.x;
  if (idx >= total) return;                 // [B,32]
  const int o = idx & 31, b = idx >> 5;
  float s = bias[o];
  for (int k = 0; k < 512; ++k) s += pooled[b * 512 + k] * w[k * 32 + o];
  semb[idx] = fmaxf(s, 0.0f);
}

// ---------------- timing MLP -----------------------------------------------
__global__ void timing1_kernel(const float* __restrict__ tv, const float* __restrict__ w,
                               const float* __restrict__ bias, float* __restrict__ o, int total)
{
  int idx = blockIdx.x * blockDim.x + threadIdx.x;
  if (idx >= total) return;                 // [B,64]
  const int j = idx & 63, b = idx >> 6;
  float s = bias[j];
  for (int k = 0; k < 16; ++k) s += tv[b * 16 + k] * w[k * 64 + j];
  o[idx] = fmaxf(s, 0.0f);
}

__global__ void timing2_kernel(const float* __restrict__ hin, const float* __restrict__ w,
                               const float* __restrict__ bias, float* __restrict__ o, int total)
{
  int idx = blockIdx.x * blockDim.x + threadIdx.x;
  if (idx >= total) return;                 // [B,64]
  const int j = idx & 63, b = idx >> 6;
  float s = bias[j];
  for (int k = 0; k < 64; ++k) s += hin[b * 64 + k] * w[k * 64 + j];
  o[idx] = fmaxf(s, 0.0f);
}

// ---------------- fused head: one block per graph --------------------------
__global__ __launch_bounds__(256)
void head_kernel(const float* __restrict__ gemb, const float* __restrict__ gs,
                 const float* __restrict__ semb, const float* __restrict__ temb,
                 const float* __restrict__ w1, const float* __restrict__ b1,
                 const float* __restrict__ w2, const float* __restrict__ b2,
                 const float* __restrict__ w3, const float* __restrict__ b3,
                 float* __restrict__ out)
{
  __shared__ float fused[224];
  __shared__ float hh1[256];
  __shared__ float hh2[128];
  const int b = blockIdx.x;
  const int t = threadIdx.x;
  if (t < 128)      fused[t] = gemb[b * 128 + t] / gs[b];
  else if (t < 160) fused[t] = semb[b * 32 + (t - 128)];
  else if (t < 224) fused[t] = temb[b * 64 + (t - 160)];
  __syncthreads();
  {
    float s = b1[t];
    for (int c = 0; c < 224; ++c) s += fused[c] * w1[c * 256 + t];
    hh1[t] = fmaxf(s, 0.0f);
  }
  __syncthreads();
  if (t < 128) {
    float s = b2[t];
    for (int c = 0; c < 256; ++c) s += hh1[c] * w2[c * 128 + t];
    hh2[t] = fmaxf(s, 0.0f);
  }
  __syncthreads();
  if (t < 3) {
    float s = b3[t];
    for (int c = 0; c < 128; ++c) s += hh2[c] * w3[c * 3 + t];
    out[b * 3 + t] = 1.0f / (1.0f + __expf(-s));
  }
}

// ===========================================================================
extern "C" void kernel_launch(void* const* d_in, const int* in_sizes, int n_in,
                              void* d_out, int out_size, void* d_ws, size_t ws_size,
                              hipStream_t stream)
{
  // ---- inputs (setup_inputs insertion order) ----
  const float* x     = (const float*)d_in[0];
  const int*   eidx  = (const int*)d_in[1];
  const float* eattr = (const float*)d_in[2];
  const int*   batch = (const int*)d_in[3];
  const float* spat  = (const float*)d_in[4];
  const float* tvec  = (const float*)d_in[5];

  const int N_ = in_sizes[0] / 64;
  const int E_ = in_sizes[1] / 2;
  const int B_ = in_sizes[4] / (128 * 128);
  const int* src = eidx;
  const int* dst = eidx + E_;

  // ---- params: jax pytree (sorted-key) flatten order, starting at d_in[6] ----
  const float* conv1_b = (const float*)d_in[6];
  const float* conv1_w = (const float*)d_in[7];
  const float* conv2_b = (const float*)d_in[8];
  const float* conv2_w = (const float*)d_in[9];
  const float* em1_b   = (const float*)d_in[10];
  const float* em1_w   = (const float*)d_in[11];
  const float* em2_b   = (const float*)d_in[12];
  const float* em2_w   = (const float*)d_in[13];
  // gat[L]: att_dst, att_src, b, w at 14 + 4L
  const float* gate1_b = (const float*)d_in[30];
  const float* gate1_w = (const float*)d_in[31];
  const float* gate2_b = (const float*)d_in[32];
  const float* gate2_w = (const float*)d_in[33];
  const float* h1_b    = (const float*)d_in[34];
  const float* h1_w    = (const float*)d_in[35];
  const float* h2_b    = (const float*)d_in[36];
  const float* h2_w    = (const float*)d_in[37];
  const float* h3_b    = (const float*)d_in[38];
  const float* h3_w    = (const float*)d_in[39];
  const float* inp_b   = (const float*)d_in[40];
  const float* inp_w   = (const float*)d_in[41];
  // norms[L]: b, g at 42 + 2L
  const float* spfc_b  = (const float*)d_in[50];
  const float* spfc_w  = (const float*)d_in[51];
  const float* t1_b    = (const float*)d_in[52];
  const float* t1_w    = (const float*)d_in[53];
  const float* t2_b    = (const float*)d_in[54];
  const float* t2_w    = (const float*)d_in[55];

  // ---- workspace bump allocator ----
  char* wsp = (char*)d_ws;
  auto alloc = [&](size_t nbytes) -> float* {
    float* p = (float*)wsp;
    wsp += (nbytes + 255) & ~(size_t)255;
    return p;
  };
  float* h     = alloc((size_t)N_ * HID_ * 4);
  float* hp    = alloc((size_t)N_ * HID_ * 4);
  float* hacc  = alloc((size_t)N_ * HID_ * 4);
  float* crit  = alloc((size_t)N_ * 4);
  float* a_s   = alloc((size_t)N_ * 4 * 4);
  float* a_d   = alloc((size_t)N_ * 4 * 4);
  float* mrow  = alloc((size_t)N_ * 4 * 4);
  float* denom = alloc((size_t)N_ * 4 * 4);
  float* gate  = alloc((size_t)N_ * 4);
  float* gm    = alloc((size_t)B_ * 4);
  float* gs    = alloc((size_t)B_ * 4);
  float* gemb  = alloc((size_t)B_ * HID_ * 4);
  float* c1o   = alloc((size_t)B_ * 16 * 64 * 64 * 4);
  float* c2o   = alloc((size_t)B_ * 32 * 32 * 32 * 4);
  float* pooled= alloc((size_t)B_ * 512 * 4);
  float* semb  = alloc((size_t)B_ * 32 * 4);
  float* thid  = alloc((size_t)B_ * 64 * 4);
  float* temb  = alloc((size_t)B_ * 64 * 4);

  const float NEG_INF = __builtin_huge_valf() * -1.0f;
  const int TB = 256;
  const int nodeWaveGrid = (int)(((long long)N_ * 32 + TB - 1) / TB);

  // ---- edge MLP -> criticality ----
  hipMemsetAsync(crit, 0, (size_t)N_ * 4, stream);
  edge_mlp_kernel<<<(E_ + TB - 1) / TB, TB, 0, stream>>>(
      eattr, dst, em1_w, em1_b, em2_w, em2_b, crit, E_);

  // ---- input projection (WMMA, K=64) ----
  gemm_n128_wmma<64><<<(N_ + 15) / 16, TB, 0, stream>>>(x, inp_w, inp_b, h, N_);

  // ---- 4 GAT layers ----
  for (int L = 0; L < 4; ++L) {
    const float* att_d = (const float*)d_in[14 + 4 * L];
    const float* att_s = (const float*)d_in[15 + 4 * L];
    const float* gat_b = (const float*)d_in[16 + 4 * L];
    const float* gat_w = (const float*)d_in[17 + 4 * L];
    const float* nrm_b = (const float*)d_in[42 + 2 * L];
    const float* nrm_g = (const float*)d_in[43 + 2 * L];

    hipMemsetAsync(hacc,  0, (size_t)N_ * HID_ * 4, stream);
    hipMemsetAsync(denom, 0, (size_t)N_ * 4 * 4, stream);
    fill_kernel<<<(N_ * 4 + TB - 1) / TB, TB, 0, stream>>>(mrow, NEG_INF, (long long)N_ * 4);

    gemm_n128_wmma<128><<<(N_ + 15) / 16, TB, 0, stream>>>(h, gat_w, nullptr, hp, N_);
    attn_coeff_kernel<<<nodeWaveGrid, TB, 0, stream>>>(hp, att_s, att_d, a_s, a_d, N_);
    edge_max_kernel<<<(E_ + N_ + TB - 1) / TB, TB, 0, stream>>>(
        src, dst, a_s, a_d, mrow, E_, N_);
    {
      long long tot = (long long)(E_ + N_) * 32;
      edge_accum_kernel<<<(int)((tot + TB - 1) / TB), TB, 0, stream>>>(
          src, dst, a_s, a_d, mrow, hp, hacc, denom, E_, N_);
    }
    node_update_kernel<<<nodeWaveGrid, TB, 0, stream>>>(
        hacc, denom, gat_b, nrm_g, nrm_b, crit, h, N_);
  }

  // ---- global attention pooling ----
  fill_kernel<<<1, TB, 0, stream>>>(gm, NEG_INF, B_);
  hipMemsetAsync(gs,   0, (size_t)B_ * 4, stream);
  hipMemsetAsync(gemb, 0, (size_t)B_ * HID_ * 4, stream);
  gate_kernel<<<nodeWaveGrid, TB, 0, stream>>>(
      h, gate1_w, gate1_b, gate2_w, gate2_b, batch, gate, gm, N_);
  pool_accum_kernel<<<nodeWaveGrid, TB, 0, stream>>>(h, gate, gm, batch, gs, gemb, N_);

  // ---- spatial encoder ----
  {
    const int t1 = B_ * 16 * 64 * 64;
    conv1_kernel<<<(t1 + TB - 1) / TB, TB, 0, stream>>>(spat, conv1_w, conv1_b, c1o, t1);
    const int t2 = B_ * 32 * 32 * 32;
    conv2_kernel<<<(t2 + TB - 1) / TB, TB, 0, stream>>>(c1o, conv2_w, conv2_b, c2o, t2);
    const int t3 = B_ * 32 * 16;
    avgpool_kernel<<<(t3 + TB - 1) / TB, TB, 0, stream>>>(c2o, pooled, t3);
    const int t4 = B_ * 32;
    sp_fc_kernel<<<(t4 + TB - 1) / TB, TB, 0, stream>>>(pooled, spfc_w, spfc_b, semb, t4);
  }

  // ---- timing MLP ----
  timing1_kernel<<<(B_ * 64 + TB - 1) / TB, TB, 0, stream>>>(tvec, t1_w, t1_b, thid, B_ * 64);
  timing2_kernel<<<(B_ * 64 + TB - 1) / TB, TB, 0, stream>>>(thid, t2_w, t2_b, temb, B_ * 64);

  // ---- fused head ----
  head_kernel<<<B_, TB, 0, stream>>>(gemb, gs, semb, temb,
                                     h1_w, h1_b, h2_w, h2_b, h3_w, h3_b,
                                     (float*)d_out);
}
